// StepwisePatchMerging_39221641347602
// MI455X (gfx1250) — matrix operand order repaired and checked by reading, not compiled, not run on hardware
//
#include <hip/hip_runtime.h>
#include <hip/hip_bf16.h>

// ---------------- WMMA types & helpers (CDNA5 / gfx1250, wave32) ----------------
typedef __attribute__((ext_vector_type(16))) __bf16 v16bf;
typedef __attribute__((ext_vector_type(8)))  float  v8f;
typedef __attribute__((ext_vector_type(4)))  int    v4i;
typedef __attribute__((ext_vector_type(4)))  float  v4f;

union U8b { v4i vi; __bf16 h[8]; };

// raw LDS byte offset of a shared-memory pointer (addrspace(3) view)
#define LDS_OFF(p) ((unsigned)(uintptr_t)(__attribute__((address_space(3))) void*)(p))

// async DMA: 16B global -> LDS, tracked by ASYNCcnt (no VGPR round trip)
__device__ __forceinline__ void async_cp16(void* lds_dst, const void* gsrc) {
  asm volatile("global_load_async_to_lds_b128 %0, %1, off"
               :: "v"(LDS_OFF(lds_dst)), "v"(gsrc) : "memory");
}
__device__ __forceinline__ void wait_async0() {
  asm volatile("s_wait_asynccnt 0x0" ::: "memory");
}

// Swizzled-operand storage: each 16x32 (A) or 32x16 (B) bf16 tile stored as
// 32 lanes x 16 contiguous bf16 (1024 B): fragment load = one aligned 32-byte
// LDS read per lane (2x ds_load_b128). Both slot functions are contiguous over
// 8 consecutive k (k0 % 8 == 0), so staging can use 16B chunks.
__device__ __forceinline__ int a_slot(int m, int k) {   // A: row m 0..15, k 0..31
  return ((k >> 4) * 16 + m) * 16 + (k & 15);
}
__device__ __forceinline__ int b_slot(int k, int n) {   // B: k 0..31, col n 0..15
  return ((k >> 4) * 16 + n) * 16 + (k & 15);
}
__device__ __forceinline__ v16bf frag(const __bf16* s, int tile) {
  return *(const v16bf*)(s + tile * 512 + (threadIdx.x & 31) * 16);
}
__device__ __forceinline__ v8f wmma_bf16(v16bf a, v16bf b, v8f c) {
  return __builtin_amdgcn_wmma_f32_16x16x32_bf16(false, a, false, b, (short)0, c,
                                                 false, false);
}
__device__ __forceinline__ v4i pack8(const v8f& a) {
  U8b u;
#pragma unroll
  for (int j = 0; j < 8; ++j) u.h[j] = (__bf16)a[j];
  return u.vi;
}
__device__ __forceinline__ v4i pack8f(v4f a, v4f b) {
  U8b u;
#pragma unroll
  for (int j = 0; j < 4; ++j) { u.h[j] = (__bf16)a[j]; u.h[4 + j] = (__bf16)b[j]; }
  return u.vi;
}

// ---------------- problem constants ----------------
static constexpr int BB  = 8;
static constexpr int CIN = 64;
static constexpr int CO  = 128;
static constexpr int HH  = 224, WW = 224;
static constexpr int HO  = 112, WO = 112;
static constexpr int LL  = HO * WO;       // 12544
static constexpr int NTOK = 5;

// Internal workspace tensors y (B,H,W,C) and g (B,Ho,Wo,C) are NHWC bf16 so all
// producers/consumers touch contiguous channel groups (16B-aligned chunks).

// =================================================================================
// Kernel 1: multi-scale depthwise conv + channel shuffle + csc 1x1 (WMMA) -> y NHWC
// LDS: xs 36864 | ds(B-swz) 32768 | as(A-swz) 16384 | yt 65536 = 151552 B
// =================================================================================
__global__ __launch_bounds__(256)
void k1_dw_csc(const float* __restrict__ x,
               const float* __restrict__ w0, const float* __restrict__ b0,
               const float* __restrict__ w1, const float* __restrict__ b1,
               const float* __restrict__ w2, const float* __restrict__ b2,
               const float* __restrict__ w3, const float* __restrict__ b3,
               const float* __restrict__ cw, const float* __restrict__ cb,
               __bf16* __restrict__ y)
{
  constexpr int TILE = 16, HALO = 4, XT = TILE + 2 * HALO;   // 24
  extern __shared__ char smem[];
  float*  xs = (float*)smem;                      // 16*24*24 f32
  __bf16* ds = (__bf16*)(smem + 36864);           // B panel 64x256, 2x16 swz tiles
  __bf16* as = (__bf16*)(smem + 69632);           // A panel 128x64, 8x2 swz tiles
  __bf16* yt = (__bf16*)(smem + 86016);           // out tile [16py*16px][128o]

  const int tid = threadIdx.x;
  const int b   = blockIdx.z;
  const int gy0 = blockIdx.y * TILE, gx0 = blockIdx.x * TILE;
  // whole staged region (incl. halo) in-bounds? -> async DMA fast path
  const bool interior = (gy0 >= HALO) && (gy0 + TILE + HALO <= HH) &&
                        (gx0 >= HALO) && (gx0 + TILE + HALO <= WW);

  // csc weights -> A-swizzled; shuffled col j <- orig (j%4)*16+j/4
  for (int idx = tid; idx < CO * 8; idx += 256) {
    int o = idx >> 3, j0 = (idx & 7) * 8;
    U8b u;
#pragma unroll
    for (int d = 0; d < 8; ++d) {
      int j = j0 + d;
      u.h[d] = (__bf16)cw[o * CIN + ((j & 3) * 16 + (j >> 2))];
    }
    *(v4i*)(as + ((o >> 4) * 2 + (j0 >> 5)) * 512 + a_slot(o & 15, j0 & 31)) = u.vi;
  }

  const float* dww[4] = {w0, w1, w2, w3};
  const float* dwb[4] = {b0, b1, b2, b3};
  const int    ksz[4] = {3, 5, 7, 9};

  for (int h = 0; h < 4; ++h) {
    __syncthreads();
    if (interior) {
      // async global->LDS: 16ch x 24 rows x 6 chunks of 16B (rows are 96B)
      for (int idx = tid; idx < 16 * XT * 6; idx += 256) {
        int cp = idx / (XT * 6);
        int r  = (idx / 6) % XT;
        int ck = idx % 6;
        async_cp16(xs + (cp * XT + r) * XT + ck * 4,
                   x + (((size_t)b * CIN + h * 16 + cp) * HH + gy0 - HALO + r) * WW +
                       gx0 - HALO + ck * 4);
      }
      wait_async0();
    } else {
      for (int idx = tid; idx < 16 * XT * XT; idx += 256) {
        int cp = idx / (XT * XT);
        int r  = (idx / XT) % XT;
        int cc = idx % XT;
        int gy = gy0 + r - HALO, gx = gx0 + cc - HALO;
        float v = 0.f;
        if (gy >= 0 && gy < HH && gx >= 0 && gx < WW)
          v = x[(((size_t)b * CIN + h * 16 + cp) * HH + gy) * WW + gx];
        xs[idx] = v;
      }
    }
    // prefetch next head's tile rows while computing this head
    if (h < 3) {
      for (int idx = tid; idx < 16 * XT; idx += 256) {
        int cp = idx / XT, r = idx % XT;
        int gy = gy0 + r - HALO;
        if (gy >= 0 && gy < HH) {
          int gx = gx0 - HALO; if (gx < 0) gx = 0;
          __builtin_prefetch(
              &x[(((size_t)b * CIN + (h + 1) * 16 + cp) * HH + gy) * WW + gx], 0, 0);
        }
      }
    }
    __syncthreads();
    const int k = ksz[h], pad = k >> 1;
    const float* wp = dww[h];
    for (int idx = tid; idx < 16 * 256; idx += 256) {
      int cp = idx >> 8, pix = idx & 255;
      int py = pix >> 4, px = pix & 15;
      float acc = dwb[h][cp];
      const float* xc = xs + cp * (XT * XT);
      for (int ky = 0; ky < k; ++ky)
        for (int kx = 0; kx < k; ++kx)
          acc += xc[(py + HALO - pad + ky) * XT + (px + HALO - pad + kx)] *
                 wp[(cp * k + ky) * k + kx];
      int j = cp * 4 + h;                      // shuffled row
      ds[((j >> 5) * 16 + (pix >> 4)) * 512 + b_slot(j & 31, pix & 15)] = (__bf16)acc;
    }
  }
  __syncthreads();

  // GEMM: (128x64)(64x256), 8 waves x 16 N-tiles, K=64 in 2 WMMA steps
  const int wave = tid >> 5, lane = tid & 31;
  const int half = lane >> 4, n = lane & 15;
  const int ob = wave * 16, o0 = ob + 8 * half;
  v4f cb0 = *(const v4f*)(cb + o0);
  v4f cb1 = *(const v4f*)(cb + o0 + 4);
  for (int nt = 0; nt < 16; ++nt) {
    v8f acc;
#pragma unroll
    for (int j = 0; j < 4; ++j) { acc[j] = cb0[j]; acc[j + 4] = cb1[j]; }
#pragma unroll
    for (int kk = 0; kk < 2; ++kk)
      acc = wmma_bf16(frag(as, wave * 2 + kk), frag(ds, kk * 16 + nt), acc);
    *(v4i*)(yt + (nt * 16 + n) * CO + o0) = pack8(acc);   // 8 consecutive channels
  }
  __syncthreads();

  // fully-coalesced NHWC write-back
  for (int idx = tid; idx < 4096; idx += 256) {
    int pos = idx >> 4, og = idx & 15;
    int py = pos >> 4, px = pos & 15;
    v4i val = *(const v4i*)(yt + pos * CO + og * 8);
    *(v4i*)(y + (((size_t)b * HH + gy0 + py) * WW + gx0 + px) * CO + og * 8) = val;
  }
}

// =================================================================================
// Kernel 2: BN(eval)+exact GELU+7x7 stride-2 depthwise conv, NHWC -> g NHWC
// LDS: act 112896 | gws_t 25088 | bnp 1536 = 139520 B
// =================================================================================
__global__ __launch_bounds__(256)
void k2_bn_gelu_ggm(const __bf16* __restrict__ y,
                    const float* __restrict__ gam, const float* __restrict__ bet,
                    const float* __restrict__ mean, const float* __restrict__ var,
                    const float* __restrict__ gw, const float* __restrict__ gb,
                    __bf16* __restrict__ g)
{
  extern __shared__ char smem[];
  __bf16* s   = (__bf16*)smem;              // [21*21][128] activated inputs
  float*  gws = (float*)(smem + 112896);    // [49][128] transposed weights
  float*  bnp = (float*)(smem + 137984);    // inv[128] | mu[128] | beta[128]

  const int tid = threadIdx.x;
  const int b   = blockIdx.z;
  const int oy0 = blockIdx.y * 8, ox0 = blockIdx.x * 8;

  if (tid < CO) {
    bnp[tid]        = gam[tid] * rsqrtf(var[tid] + 1e-5f);
    bnp[CO + tid]   = mean[tid];
    bnp[2*CO + tid] = bet[tid];
  }
  for (int idx = tid; idx < 49 * CO; idx += 256) {
    int c = idx & 127, tap = idx >> 7;
    gws[tap * CO + c] = gw[c * 49 + tap];
  }
  __syncthreads();

  const int iy0 = oy0 * 2 - 3, ix0 = ox0 * 2 - 3;
  for (int idx = tid; idx < 441 * 16; idx += 256) {   // 21x21 pos x 16 ch-groups
    int pos = idx >> 4, c0 = (idx & 15) * 8;
    int r = pos / 21, cc = pos % 21;
    int gy = iy0 + r, gx = ix0 + cc;
    U8b u;
    if (gy >= 0 && gy < HH && gx >= 0 && gx < WW)
      u.vi = *(const v4i*)(y + (((size_t)b * HH + gy) * WW + gx) * CO + c0);
    else { v4i z = {0, 0, 0, 0}; u.vi = z; }
    U8b o;
#pragma unroll
    for (int j = 0; j < 8; ++j) {
      float v = (float)u.h[j];
      v = (v - bnp[CO + c0 + j]) * bnp[c0 + j] + bnp[2*CO + c0 + j];
      v = 0.5f * v * (1.f + erff(v * 0.70710678118654752440f));   // exact GELU
      o.h[j] = (__bf16)v;
    }
    *(v4i*)(s + pos * CO + c0) = o.vi;
  }
  __syncthreads();

  for (int idx = tid; idx < 64 * 16; idx += 256) {    // 8x8 out x 16 ch-groups
    int pos = idx >> 4, c0 = (idx & 15) * 8;
    int oy = pos >> 3, ox = pos & 7;
    v4f g0 = *(const v4f*)(gb + c0);
    v4f g1 = *(const v4f*)(gb + c0 + 4);
    float acc[8];
#pragma unroll
    for (int j = 0; j < 4; ++j) { acc[j] = g0[j]; acc[4 + j] = g1[j]; }
#pragma unroll
    for (int ky = 0; ky < 7; ++ky)
#pragma unroll
      for (int kx = 0; kx < 7; ++kx) {
        U8b xv;
        xv.vi = *(const v4i*)(s + ((oy * 2 + ky) * 21 + ox * 2 + kx) * CO + c0);
        v4f wl = *(const v4f*)(gws + (ky * 7 + kx) * CO + c0);
        v4f wh = *(const v4f*)(gws + (ky * 7 + kx) * CO + c0 + 4);
#pragma unroll
        for (int j = 0; j < 4; ++j) {
          acc[j]     += (float)xv.h[j]     * wl[j];
          acc[4 + j] += (float)xv.h[4 + j] * wh[j];
        }
      }
    U8b o;
#pragma unroll
    for (int j = 0; j < 8; ++j) o.h[j] = (__bf16)acc[j];
    *(v4i*)(g + (((size_t)b * HO + oy0 + oy) * WO + ox0 + ox) * CO + c0) = o.vi;
  }
}

// token staging via async DMA (pure bf16 copy into swizzled panel); caller must
// wait_async0() before the barrier. col n = p*16+lw, p=0 guide, p>0 unfold(2x2)
__device__ __forceinline__ void stage_tokens(const __bf16* y, const __bf16* g,
                                             __bf16* Ts, int b, int l0) {
  for (int idx = threadIdx.x; idx < 80 * 16; idx += 256) {
    int pos = idx >> 4, c0 = (idx & 15) * 8;
    int p = pos >> 4, lw = pos & 15;
    int l = l0 + lw, ho = l / WO, wo = l % WO;
    const __bf16* src;
    if (p == 0) {
      src = g + (((size_t)b * HO + ho) * WO + wo) * CO + c0;
    } else {
      int t = p - 1;
      src = y + (((size_t)b * HH + ho * 2 + (t >> 1)) * WW + wo * 2 + (t & 1)) * CO + c0;
    }
    async_cp16(Ts + ((c0 >> 5) * 5 + p) * 512 + b_slot(c0 & 31, lw), src);
  }
}
// stage one 128x128 f32 weight matrix into A-swizzled bf16 tiles (8x4)
__device__ __forceinline__ void stage_wmat(const float* w, __bf16* A) {
  for (int idx = threadIdx.x; idx < CO * 16; idx += 256) {
    int o = idx >> 4, c0 = (idx & 15) * 8;
    v4f lo = *(const v4f*)(w + o * CO + c0);
    v4f hi = *(const v4f*)(w + o * CO + c0 + 4);
    *(v4i*)(A + ((o >> 4) * 4 + (c0 >> 5)) * 512 + a_slot(o & 15, c0 & 31)) =
        pack8f(lo, hi);
  }
}

// =================================================================================
// Kernel 3: tokens -> Q(p=0), K (WMMA) -> partial attn logits (p=0 row) -> atomics
// LDS: Ts 20480 | Aq 32768 | Ak 32768 | Qs_t 8192 | Ks_t 40960 = 135168 B
// =================================================================================
__global__ __launch_bounds__(256)
void k3_attn_logits(const __bf16* __restrict__ y, const __bf16* __restrict__ g,
                    const float* __restrict__ qkvw, const float* __restrict__ qkvb,
                    float* __restrict__ S)
{
  extern __shared__ char smem[];
  __bf16* Ts  = (__bf16*)smem;
  __bf16* Aq  = (__bf16*)(smem + 20480);
  __bf16* Ak  = (__bf16*)(smem + 53248);
  float*  Qst = (float*)(smem + 86016);     // [16 n][128 o]
  float*  Kst = (float*)(smem + 94208);     // [80 n][128 o]

  const int tid = threadIdx.x;
  const int b  = blockIdx.y;
  const int l0 = blockIdx.x * 16;

  stage_tokens(y, g, Ts, b, l0);            // async DMA
  stage_wmat(qkvw, Aq);
  stage_wmat(qkvw + CO * CO, Ak);
  wait_async0();
  __syncthreads();

  const int wave = tid >> 5, lane = tid & 31;
  const int half = lane >> 4, n = lane & 15;
  const int ob = wave * 16, o0 = ob + 8 * half;

  {   // Q, token p=0 only
    v4f bl = *(const v4f*)(qkvb + o0);
    v4f bh = *(const v4f*)(qkvb + o0 + 4);
    v8f acc;
#pragma unroll
    for (int j = 0; j < 4; ++j) { acc[j] = bl[j]; acc[j + 4] = bh[j]; }
#pragma unroll
    for (int kk = 0; kk < 4; ++kk)
      acc = wmma_bf16(frag(Aq, wave * 4 + kk), frag(Ts, kk * 5 + 0), acc);
    v4f lo, hi;
#pragma unroll
    for (int j = 0; j < 4; ++j) { lo[j] = acc[j]; hi[j] = acc[j + 4]; }
    *(v4f*)(Qst + n * CO + o0) = lo;
    *(v4f*)(Qst + n * CO + o0 + 4) = hi;
  }
  v4f bl = *(const v4f*)(qkvb + CO + o0);
  v4f bh = *(const v4f*)(qkvb + CO + o0 + 4);
  for (int nt = 0; nt < NTOK; ++nt) {   // K, all 5 tokens
    v8f acc;
#pragma unroll
    for (int j = 0; j < 4; ++j) { acc[j] = bl[j]; acc[j + 4] = bh[j]; }
#pragma unroll
    for (int kk = 0; kk < 4; ++kk)
      acc = wmma_bf16(frag(Ak, wave * 4 + kk), frag(Ts, kk * 5 + nt), acc);
    v4f lo, hi;
#pragma unroll
    for (int j = 0; j < 4; ++j) { lo[j] = acc[j]; hi[j] = acc[j + 4]; }
    *(v4f*)(Kst + (nt * 16 + n) * CO + o0) = lo;
    *(v4f*)(Kst + (nt * 16 + n) * CO + o0 + 4) = hi;
  }
  __syncthreads();

  for (int idx = tid; idx < CO * NTOK; idx += 256) {
    int o = idx / NTOK, q = idx % NTOK;
    float s = 0.f;
#pragma unroll
    for (int lw = 0; lw < 16; ++lw)
      s += Qst[lw * CO + o] * Kst[(q * 16 + lw) * CO + o];
    atomicAdd(&S[((size_t)b * CO + o) * NTOK + q], s);
  }
}

// =================================================================================
// Kernel 4: 5-way softmax with 1/sqrt(Co) scale -> W5[b][o][q]
// =================================================================================
__global__ void k4_softmax(const float* __restrict__ S, float* __restrict__ W5)
{
  int i = blockIdx.x * blockDim.x + threadIdx.x;
  if (i >= BB * CO) return;
  const float sc = 0.08838834764831845f;   // 1/sqrt(128)
  float v[NTOK], m = -1e30f;
#pragma unroll
  for (int q = 0; q < NTOK; ++q) { v[q] = S[i * NTOK + q] * sc; m = fmaxf(m, v[q]); }
  float sum = 0.f;
#pragma unroll
  for (int q = 0; q < NTOK; ++q) { v[q] = expf(v[q] - m); sum += v[q]; }
  float inv = 1.f / sum;
#pragma unroll
  for (int q = 0; q < NTOK; ++q) W5[i * NTOK + q] = v[q] * inv;
}

// =================================================================================
// Kernel 5: tokens -> V (WMMA) -> attn-weighted combine -> proj (WMMA) -> out NCHW
// LDS: Ts 20480 | Av 32768 | Ap 32768 | Vs_t 40960 | Us 4096 | ot 32768 | W5 2560
// =================================================================================
__global__ __launch_bounds__(256)
void k5_out(const __bf16* __restrict__ y, const __bf16* __restrict__ g,
            const float* __restrict__ qkvw, const float* __restrict__ qkvb,
            const float* __restrict__ W5,
            const float* __restrict__ pw, const float* __restrict__ pb,
            float* __restrict__ out)
{
  extern __shared__ char smem[];
  __bf16* Ts  = (__bf16*)smem;
  __bf16* Av  = (__bf16*)(smem + 20480);
  __bf16* Ap  = (__bf16*)(smem + 53248);
  float*  Vst = (float*)(smem + 86016);     // [80 n][128 o]
  __bf16* Us  = (__bf16*)(smem + 126976);   // K=128 x N=16, 4 swz tiles
  float*  ot  = (float*)(smem + 131072);    // [128 oo][16 n]
  float*  Ws5 = (float*)(smem + 163840);    // [128][5]

  const int tid = threadIdx.x;
  const int b  = blockIdx.y;
  const int l0 = blockIdx.x * 16;

  stage_tokens(y, g, Ts, b, l0);            // async DMA
  if (tid < 160)                            // 2.5KB W5 copy via async DMA
    async_cp16(Ws5 + tid * 4, W5 + (size_t)b * CO * NTOK + tid * 4);
  stage_wmat(qkvw + 2 * CO * CO, Av);
  stage_wmat(pw, Ap);
  wait_async0();
  __syncthreads();

  const int wave = tid >> 5, lane = tid & 31;
  const int half = lane >> 4, n = lane & 15;
  const int ob = wave * 16, o0 = ob + 8 * half;

  // V = Wv * T (bias folded into combine since sum_q attn = 1)
  for (int nt = 0; nt < NTOK; ++nt) {
    v8f acc = {};
#pragma unroll
    for (int kk = 0; kk < 4; ++kk)
      acc = wmma_bf16(frag(Av, wave * 4 + kk), frag(Ts, kk * 5 + nt), acc);
    v4f lo, hi;
#pragma unroll
    for (int j = 0; j < 4; ++j) { lo[j] = acc[j]; hi[j] = acc[j + 4]; }
    *(v4f*)(Vst + (nt * 16 + n) * CO + o0) = lo;
    *(v4f*)(Vst + (nt * 16 + n) * CO + o0 + 4) = hi;
  }
  __syncthreads();

  // u[o,lw] = bv[o] + sum_q W5[o,q] * V[o, q*16+lw]  -> B-swizzled for proj GEMM
  {
    int lw = tid >> 4, c0 = (tid & 15) * 8;
    v4f bvl = *(const v4f*)(qkvb + 2 * CO + c0);
    v4f bvh = *(const v4f*)(qkvb + 2 * CO + c0 + 4);
    float s[8];
#pragma unroll
    for (int j = 0; j < 4; ++j) { s[j] = bvl[j]; s[4 + j] = bvh[j]; }
#pragma unroll
    for (int q = 0; q < NTOK; ++q) {
      v4f vl = *(const v4f*)(Vst + (q * 16 + lw) * CO + c0);
      v4f vh = *(const v4f*)(Vst + (q * 16 + lw) * CO + c0 + 4);
#pragma unroll
      for (int j = 0; j < 4; ++j) {
        s[j]     += Ws5[(c0 + j) * NTOK + q] * vl[j];
        s[4 + j] += Ws5[(c0 + 4 + j) * NTOK + q] * vh[j];
      }
    }
    U8b u;
#pragma unroll
    for (int j = 0; j < 8; ++j) u.h[j] = (__bf16)s[j];
    *(v4i*)(Us + (c0 >> 5) * 512 + b_slot(c0 & 31, lw)) = u.vi;
  }
  __syncthreads();

  // proj: (128x128)(128x16) -> ot tile
  {
    v4f bl = *(const v4f*)(pb + o0);
    v4f bh = *(const v4f*)(pb + o0 + 4);
    v8f acc;
#pragma unroll
    for (int j = 0; j < 4; ++j) { acc[j] = bl[j]; acc[j + 4] = bh[j]; }
#pragma unroll
    for (int kk = 0; kk < 4; ++kk)
      acc = wmma_bf16(frag(Ap, wave * 4 + kk), frag(Us, kk), acc);
#pragma unroll
    for (int j = 0; j < 8; ++j) ot[(o0 + j) * 16 + n] = acc[j];
  }
  __syncthreads();

  // coalesced b128 stores into NCHW fp32 output (L-chunks 16-aligned, 112%16==0)
  for (int idx = tid; idx < 512; idx += 256) {
    int oo = idx >> 2, q4 = idx & 3;
    v4f val = *(const v4f*)(ot + oo * 16 + q4 * 4);
    *(v4f*)(out + ((size_t)b * CO + oo) * LL + l0 + q4 * 4) = val;
  }
}

// =================================================================================
extern "C" void kernel_launch(void* const* d_in, const int* in_sizes, int n_in,
                              void* d_out, int out_size, void* d_ws, size_t ws_size,
                              hipStream_t stream)
{
  const float* x    = (const float*)d_in[0];
  const float* dw_w0 = (const float*)d_in[1];  const float* dw_b0 = (const float*)d_in[2];
  const float* dw_w1 = (const float*)d_in[3];  const float* dw_b1 = (const float*)d_in[4];
  const float* dw_w2 = (const float*)d_in[5];  const float* dw_b2 = (const float*)d_in[6];
  const float* dw_w3 = (const float*)d_in[7];  const float* dw_b3 = (const float*)d_in[8];
  const float* csc_w = (const float*)d_in[9];  const float* csc_b = (const float*)d_in[10];
  const float* bn_g  = (const float*)d_in[11]; const float* bn_b  = (const float*)d_in[12];
  const float* bn_m  = (const float*)d_in[13]; const float* bn_v  = (const float*)d_in[14];
  const float* ggm_w = (const float*)d_in[15]; const float* ggm_b = (const float*)d_in[16];
  const float* qkv_w = (const float*)d_in[17]; const float* qkv_b = (const float*)d_in[18];
  const float* pw    = (const float*)d_in[19]; const float* pbias = (const float*)d_in[20];

  char* ws = (char*)d_ws;
  const size_t YB = (size_t)BB * CO * HH * WW * sizeof(__bf16);   // ~102.8 MB
  const size_t GB = (size_t)BB * CO * HO * WO * sizeof(__bf16);   // ~25.7 MB
  __bf16* y  = (__bf16*)ws;                                       // NHWC
  __bf16* g  = (__bf16*)(ws + YB);                                // NHWC
  float*  S  = (float*)(ws + YB + GB);
  float*  W5 = (float*)(ws + YB + GB + (size_t)BB * CO * NTOK * sizeof(float));

  hipMemsetAsync(S, 0, (size_t)BB * CO * NTOK * sizeof(float), stream);

  dim3 g1(WW / 16, HH / 16, BB);                 // 14 x 14 x 8
  k1_dw_csc<<<g1, 256, 151552, stream>>>(x, dw_w0, dw_b0, dw_w1, dw_b1,
                                         dw_w2, dw_b2, dw_w3, dw_b3,
                                         csc_w, csc_b, y);

  dim3 g2(WO / 8, HO / 8, BB);                   // 14 x 14 x 8
  k2_bn_gelu_ggm<<<g2, 256, 139520, stream>>>(y, bn_g, bn_b, bn_m, bn_v,
                                              ggm_w, ggm_b, g);

  dim3 g3(LL / 16, BB);                          // 784 x 8
  k3_attn_logits<<<g3, 256, 135168, stream>>>(y, g, qkv_w, qkv_b, S);

  k4_softmax<<<(BB * CO + 255) / 256, 256, 0, stream>>>(S, W5);

  dim3 g5(LL / 16, BB);
  k5_out<<<g5, 256, 166400, stream>>>(y, g, qkv_w, qkv_b, W5, pw, pbias,
                                      (float*)d_out);
}
